// TransformerEncoder_57621281243565
// MI455X (gfx1250) — compile-verified, matrix operand order checked by reading, not run on hardware
//
#include <hip/hip_runtime.h>
#include <hip/hip_bf16.h>
#include <stdint.h>

// ---------------------------------------------------------------------------
// Types for CDNA5 WMMA (wave32): A/B = 16 halfs (8 VGPRs), C/D = 8 floats.
// ---------------------------------------------------------------------------
typedef _Float16 v16h __attribute__((ext_vector_type(16)));
typedef float    v8f  __attribute__((ext_vector_type(8)));

#define BB   64
#define NN   32
#define DD   32
#define HH   8
#define HD   256
#define MNOD (BB * NN)            // 2048 node rows
#define MEDG (BB * NN * NN)       // 65536 edge rows
#define INV_SQRT_D 0.17677669529663689f

// gfx1250 async global->LDS path (ASYNCcnt), if the toolchain exposes it.
#if defined(__has_builtin)
#if __has_builtin(__builtin_amdgcn_global_load_async_to_lds_b128) && \
    __has_builtin(__builtin_amdgcn_s_wait_asynccnt)
#define HAVE_ASYNC_LDS 1
#endif
#endif
#ifndef HAVE_ASYNC_LDS
#define HAVE_ASYNC_LDS 0
#endif

#if HAVE_ASYNC_LDS
// Parameter types per hipcc diagnostic: pointer to 16-byte int vector,
// AS1 (__device__) for the global source, AS3 (__shared__) for the LDS dest.
typedef int i32x4 __attribute__((vector_size(4 * sizeof(int))));
typedef __attribute__((address_space(1))) i32x4 glb_i32x4;
typedef __attribute__((address_space(3))) i32x4 lds_i32x4;
#endif

static __device__ __forceinline__ v8f wmma16(v16h a, v16h b, v8f c) {
  // D = A(16x32,f16) * B(32x16,f16) + C(16x16,f32)
  return __builtin_amdgcn_wmma_f32_16x16x32_f16(
      /*neg_a=*/false, a, /*neg_b=*/false, b,
      /*c_mod=*/(short)0, c, /*reuse_a=*/false, /*reuse_b=*/false);
}

// Build a 16-bit A/B fragment from a row-major fp32 row (ISA 7.12.2 layout:
// lanes 0-15: halfs[0..7]=K[kb..kb+7], halfs[8..15]=K[kb+16..kb+23];
// lanes 16-31: +8 on K).  rowp points at element (row, kb + g*8).
static __device__ __forceinline__ v16h frag_f32(const float* __restrict__ rowp) {
  v16h r;
#pragma unroll
  for (int t = 0; t < 8; ++t) {
    r[t]     = (_Float16)rowp[t];
    r[t + 8] = (_Float16)rowp[t + 16];
  }
  return r;
}
static __device__ __forceinline__ v16h frag_f16(const _Float16* __restrict__ rowp) {
  v16h r;
#pragma unroll
  for (int t = 0; t < 8; ++t) {
    r[t]     = rowp[t];
    r[t + 8] = rowp[t + 16];
  }
  return r;
}

// ---------------------------------------------------------------------------
// Weight prep: W[K][N] fp32  ->  Wt[N][K] f16 (contiguous per-lane B frags)
// ---------------------------------------------------------------------------
__global__ void k_prep_wt(const float* __restrict__ W, _Float16* __restrict__ Wt,
                          int K, int N) {
  int i = blockIdx.x * blockDim.x + threadIdx.x;
  if (i >= K * N) return;
  int k = i / N, n = i % N;
  Wt[(size_t)n * K + k] = (_Float16)W[i];
}

// ---------------------------------------------------------------------------
// Small-K linear + ReLU (K = 8/16 input encoders): plain VALU.
// ---------------------------------------------------------------------------
__global__ void k_lin_relu(const float* __restrict__ X, const float* __restrict__ W,
                           const float* __restrict__ b, float* __restrict__ Y,
                           int M, int N, int K) {
  int i = blockIdx.x * blockDim.x + threadIdx.x;
  if (i >= M * N) return;
  int m = i / N, n = i % N;
  float acc = b[n];
  for (int k = 0; k < K; ++k) acc += X[(size_t)m * K + k] * W[(size_t)k * N + n];
  Y[i] = fmaxf(acc, 0.0f);
}

// ---------------------------------------------------------------------------
// Generic WMMA GEMM: C[M][N] = act(A[M][K](f32) * Wt[N][K](f16)^T + bias)
// One wave per 16x16 output tile.  M,N multiples of 16; K multiple of 32.
// ---------------------------------------------------------------------------
__global__ __launch_bounds__(32)
void k_gemm(const float* __restrict__ A, const _Float16* __restrict__ Wt,
            const float* __restrict__ bias, float* __restrict__ C,
            int M, int N, int K, int relu) {
  int lane = threadIdx.x & 31;
  int g = lane >> 4, idx = lane & 15;
  int m0 = blockIdx.x * 16, n0 = blockIdx.y * 16;

  const float*    arow = A  + (size_t)(m0 + idx) * K + g * 8;
  const _Float16* brow = Wt + (size_t)(n0 + idx) * K + g * 8;

  v8f acc = {};
  for (int kc = 0; kc < K; kc += 32) {
    if (kc + 64 < K) __builtin_prefetch(arow + kc + 64, 0, 1);  // global_prefetch_b8
    v16h af = frag_f32(arow + kc);
    v16h bf = frag_f16(brow + kc);
    acc = wmma16(af, bf, acc);
  }

  int col = n0 + idx;
  float bv = bias[col];
#pragma unroll
  for (int r = 0; r < 8; ++r) {
    int row = m0 + r + 8 * g;             // C/D layout: VGPR r -> M = r + 8*g
    float v = acc[r] + bv;
    if (relu) v = fmaxf(v, 0.0f);
    C[(size_t)row * N + col] = v;
  }
}

// ---------------------------------------------------------------------------
// Fused attention block: per (b, 16-row i-tile), 8 waves.
// K/V for batch b staged once into LDS (async-to-LDS when available);
// per i: em/ea WMMA -> attn elementwise -> new_edges in LDS -> scores ->
// softmax -> wv, and eo-projection (WMMA out of LDS) -> ae.
// em/ea/edges_flat are NEVER materialized in global memory.
// ---------------------------------------------------------------------------
__global__ __launch_bounds__(256)
void k_attn(const float* __restrict__ edges,     // [B][N][N][D]
            const float* __restrict__ q,         // [B*N][HD]
            const float* __restrict__ kmat,      // [B*N][HD]
            const float* __restrict__ vmat,      // [B*N][HD]
            const _Float16* __restrict__ wmul_t, // [HD][D]
            const float* __restrict__ bmul,      // [HD]
            const _Float16* __restrict__ wadd_t, // [HD][D]
            const float* __restrict__ badd,      // [HD]
            const _Float16* __restrict__ weo_t,  // [D][HD]
            const float* __restrict__ beo,       // [D]
            float* __restrict__ wvout,           // [B*N][HD]
            float* __restrict__ aeout) {         // [B*N*N][D]
  __shared__ __align__(16) float ne[NN][HD];    // new_edges tile for one i (32 KB)
  __shared__ __align__(16) float kls[NN][HD];   // K[b] staged (32 KB)
  __shared__ __align__(16) float vls[NN][HD];   // V[b] staged (32 KB)
  __shared__ float sc[HH][NN];
  __shared__ float ex[HH][NN];
  __shared__ float pw[HH][NN];

  int b = blockIdx.x, it = blockIdx.y;
  int tid = threadIdx.x;
  int wave = tid >> 5, lane = tid & 31;
  int g = lane >> 4, idx = lane & 15;

  // ---- Stage K[b], V[b] into LDS (8192 floats each = 2048 b128 chunks) ----
  {
    const float* kg = kmat + (size_t)b * NN * HD;
    const float* vg = vmat + (size_t)b * NN * HD;
    float* kl = &kls[0][0];
    float* vl = &vls[0][0];
#if HAVE_ASYNC_LDS
    for (int c = tid; c < (NN * HD) / 4; c += 256) {
      // generic LDS address low 32 bits == LDS offset (ISA 10.2 aperture calc)
      __builtin_amdgcn_global_load_async_to_lds_b128(
          (glb_i32x4*)(uintptr_t)(kg + c * 4),
          (lds_i32x4*)(uint32_t)(uintptr_t)(kl + c * 4), 0, 0);
      __builtin_amdgcn_global_load_async_to_lds_b128(
          (glb_i32x4*)(uintptr_t)(vg + c * 4),
          (lds_i32x4*)(uint32_t)(uintptr_t)(vl + c * 4), 0, 0);
    }
    __builtin_amdgcn_s_wait_asynccnt(0);
#else
    for (int c = tid; c < (NN * HD) / 4; c += 256) {
      float4 kd = *(const float4*)(kg + c * 4);
      float4 vd = *(const float4*)(vg + c * 4);
      *(float4*)(kl + c * 4) = kd;
      *(float4*)(vl + c * 4) = vd;
    }
#endif
  }
  __syncthreads();

  for (int ii = 0; ii < 16; ++ii) {
    int i = it * 16 + ii;
    const float* qrow = q + ((size_t)b * NN + i) * HD;

    // ---- Phase 1: em/ea GEMM + attn elementwise -> ne[j][hd] in LDS ----
    {
      int rt = wave >> 2;                 // j row-tile (0/1)
      int cbase = (wave & 3) * 64;        // 4 col-tiles of 16 per wave
      const float* arow =
          edges + ((((size_t)b * NN + i) * NN) + rt * 16 + idx) * DD + g * 8;
      v16h af = frag_f32(arow);           // K = 32 (single chunk)
#pragma unroll
      for (int t = 0; t < 4; ++t) {
        int n0 = cbase + t * 16;
        v16h bm = frag_f16(wmul_t + (size_t)(n0 + idx) * DD + g * 8);
        v16h ba = frag_f16(wadd_t + (size_t)(n0 + idx) * DD + g * 8);
        v8f cm = {}, ca = {};
        cm = wmma16(af, bm, cm);
        ca = wmma16(af, ba, ca);
        int col = n0 + idx;
        float qv = qrow[col] * INV_SQRT_D;
#pragma unroll
        for (int r = 0; r < 8; ++r) {
          int j = rt * 16 + r + 8 * g;
          float a  = qv * kls[j][col];
          float em = cm[r] + bmul[col];
          float ea = ca[r] + badd[col];
          ne[j][col] = em * a + ea + a;   // new_edges
        }
      }
    }
    __syncthreads();

    // ---- Phase 2: scores[h][j] = sum_d ne[j][h*D+d] ----
    {
      int h = tid >> 5, j = tid & 31;
      float s = 0.0f;
      for (int d = 0; d < DD; ++d) s += ne[j][h * DD + d];
      sc[h][j] = s;
    }
    __syncthreads();

    // ---- Phase 3: softmax over j ----
    {
      int h = tid >> 5, j = tid & 31;
      float m = -3.4e38f;
      for (int jj = 0; jj < NN; ++jj) m = fmaxf(m, sc[h][jj]);
      ex[h][j] = __expf(sc[h][j] - m);
    }
    __syncthreads();
    {
      int h = tid >> 5, j = tid & 31;
      float s = 0.0f;
      for (int jj = 0; jj < NN; ++jj) s += ex[h][jj];
      pw[h][j] = ex[h][j] / s;
    }
    __syncthreads();

    // ---- Phase 4: wv row (thread = hd column), V from LDS ----
    {
      int col = tid, h = col >> 5;
      float acc = 0.0f;
      for (int j = 0; j < NN; ++j) acc += pw[h][j] * vls[j][col];
      wvout[((size_t)b * NN + i) * HD + col] = acc;
    }

    // ---- Phase 5: ae[i] = ne(32x256) @ Weo + beo  (waves 0..3, WMMA) ----
    if (wave < 4) {
      int rt = wave >> 1;   // j row-tile
      int ct = wave & 1;    // output col-tile
      v8f acc = {};
      for (int kc = 0; kc < HD; kc += 32) {
        v16h af;
#pragma unroll
        for (int t = 0; t < 8; ++t) {   // A frag straight out of LDS
          af[t]     = (_Float16)ne[rt * 16 + idx][kc + g * 8 + t];
          af[t + 8] = (_Float16)ne[rt * 16 + idx][kc + g * 8 + 16 + t];
        }
        v16h bf = frag_f16(weo_t + (size_t)(ct * 16 + idx) * HD + kc + g * 8);
        acc = wmma16(af, bf, acc);
      }
      int n = ct * 16 + idx;
      float bv = beo[n];
#pragma unroll
      for (int r = 0; r < 8; ++r) {
        int j = rt * 16 + r + 8 * g;
        aeout[(((size_t)b * NN + i) * NN + j) * DD + n] = acc[r] + bv;
      }
    }
    __syncthreads();  // protect ne before next i iteration
  }
}

// ---------------------------------------------------------------------------
// out[b] = LayerNorm(x[b] + res[b]) over `per` elements (gamma=1, beta=0).
// ---------------------------------------------------------------------------
__global__ __launch_bounds__(256)
void k_add_ln(const float* __restrict__ x, const float* __restrict__ res,
              float* __restrict__ out, int per) {
  __shared__ float red[256];
  __shared__ float stats[2];
  size_t base = (size_t)blockIdx.x * (size_t)per;
  int tid = threadIdx.x;

  float s = 0.0f;
  for (int t = tid; t < per; t += 256) s += x[base + t] + res[base + t];
  red[tid] = s; __syncthreads();
  for (int o = 128; o > 0; o >>= 1) { if (tid < o) red[tid] += red[tid + o]; __syncthreads(); }
  if (tid == 0) stats[0] = red[0] / (float)per;
  __syncthreads();
  float mean = stats[0];

  float v = 0.0f;
  for (int t = tid; t < per; t += 256) {
    float d = x[base + t] + res[base + t] - mean;
    v += d * d;
  }
  red[tid] = v; __syncthreads();
  for (int o = 128; o > 0; o >>= 1) { if (tid < o) red[tid] += red[tid + o]; __syncthreads(); }
  if (tid == 0) stats[1] = rsqrtf(red[0] / (float)per + 1e-5f);
  __syncthreads();
  float rstd = stats[1];

  for (int t = tid; t < per; t += 256)
    out[base + t] = (x[base + t] + res[base + t] - mean) * rstd;
}

// ---------------------------------------------------------------------------
// Host orchestration.
// Params assume jax-style pytree flattening (sorted dict keys):
//  0 edges, 1 nodes, 2/3 in_e1.{b,w}, 4/5 in_e2, 6/7 in_n1, 8/9 in_n2,
//  per layer l (base = 10 + 30*l): add.{b,w}, eo, k, mul, no, q, v (attn),
//    ln_e1_b/w, ln_e2_b/w, ln_n1_b/w, ln_n2_b/w (identity), me1, me2, mn1, mn2
//  70..77: mm1.{b,w}, mm2, mv1, mv2
// ---------------------------------------------------------------------------
extern "C" void kernel_launch(void* const* d_in, const int* in_sizes, int n_in,
                              void* d_out, int out_size, void* d_ws, size_t ws_size,
                              hipStream_t stream) {
  (void)in_sizes; (void)n_in; (void)out_size; (void)ws_size;
  const float* edges_in = (const float*)d_in[0];
  const float* nodes_in = (const float*)d_in[1];
  auto P = [&](int i) { return (const float*)d_in[i]; };

  // ---- carve workspace: f16 transposed weights first, then fp32 buffers ----
  _Float16* wt_cur = (_Float16*)d_ws;
  auto prep = [&](const float* W, int K, int N) -> _Float16* {
    _Float16* dst = wt_cur;
    wt_cur += (size_t)K * N;
    int tot = K * N;
    k_prep_wt<<<dim3((tot + 255) / 256), dim3(256), 0, stream>>>(W, dst, K, N);
    return dst;
  };

  _Float16 *wt_q[2], *wt_k[2], *wt_v[2], *wt_mul[2], *wt_add[2], *wt_no[2],
           *wt_eo[2], *wt_mn1[2], *wt_mn2[2], *wt_me1[2], *wt_me2[2];
  for (int l = 0; l < 2; ++l) {
    int base = 10 + 30 * l;
    wt_add[l] = prep(P(base + 1),  DD, HD);
    wt_eo[l]  = prep(P(base + 3),  HD, DD);
    wt_k[l]   = prep(P(base + 5),  DD, HD);
    wt_mul[l] = prep(P(base + 7),  DD, HD);
    wt_no[l]  = prep(P(base + 9),  HD, DD);
    wt_q[l]   = prep(P(base + 11), DD, HD);
    wt_v[l]   = prep(P(base + 13), DD, HD);
    wt_me1[l] = prep(P(base + 23), DD, DD);
    wt_me2[l] = prep(P(base + 25), DD, DD);
    wt_mn1[l] = prep(P(base + 27), DD, DD);
    wt_mn2[l] = prep(P(base + 29), DD, DD);
  }
  _Float16* wt_in_n2 = prep(P(9),  64, 32);
  _Float16* wt_in_e2 = prep(P(5),  64, 32);
  _Float16* wt_mm1   = prep(P(71), 1024, 512);
  _Float16* wt_mm2   = prep(P(73), 512, 256);
  _Float16* wt_mv1   = prep(P(75), 1024, 512);
  _Float16* wt_mv2   = prep(P(77), 512, 256);

  float* f = (float*)((((uintptr_t)wt_cur) + 255) & ~(uintptr_t)255);
  auto falloc = [&](size_t n) { float* p = f; f += n; return p; };
  float* nodesA = falloc((size_t)MNOD * DD);
  float* nodesB = falloc((size_t)MNOD * DD);
  float* edgesA = falloc((size_t)MEDG * DD);
  float* edgesB = falloc((size_t)MEDG * DD);
  float* qb     = falloc((size_t)MNOD * HD);
  float* kb     = falloc((size_t)MNOD * HD);
  float* vb     = falloc((size_t)MNOD * HD);
  float* wvb    = falloc((size_t)MNOD * HD);
  float* anb    = falloc((size_t)MNOD * DD);
  float* tnb    = falloc((size_t)MNOD * DD);
  float* big    = falloc((size_t)MEDG * 64);   // edge tmp64 / ae+te
  float* aeb    = big;
  float* teb    = big + (size_t)MEDG * DD;

  auto gemm = [&](const float* A, const _Float16* Wt, const float* bias,
                  float* C, int M, int N, int K, int relu) {
    k_gemm<<<dim3(M / 16, N / 16), dim3(32), 0, stream>>>(A, Wt, bias, C, M, N, K, relu);
  };

  // ---- input encoders ----
  k_lin_relu<<<dim3((MNOD * 64 + 255) / 256), dim3(256), 0, stream>>>(
      nodes_in, P(7), P(6), qb, MNOD, 64, 16);          // qb as 2048x64 tmp
  gemm(qb, wt_in_n2, P(8), nodesA, MNOD, 32, 64, 1);
  k_lin_relu<<<dim3((MEDG * 64 + 255) / 256), dim3(256), 0, stream>>>(
      edges_in, P(3), P(2), big, MEDG, 64, 8);
  gemm(big, wt_in_e2, P(4), edgesA, MEDG, 32, 64, 1);

  // ---- transformer layers ----
  for (int l = 0; l < 2; ++l) {
    int base = 10 + 30 * l;
    gemm(nodesA, wt_q[l], P(base + 10), qb, MNOD, HD, DD, 0);
    gemm(nodesA, wt_k[l], P(base + 4),  kb, MNOD, HD, DD, 0);
    gemm(nodesA, wt_v[l], P(base + 12), vb, MNOD, HD, DD, 0);

    k_attn<<<dim3(BB, 2), dim3(256), 0, stream>>>(
        edgesA, qb, kb, vb,
        wt_mul[l], P(base + 6), wt_add[l], P(base + 0),
        wt_eo[l],  P(base + 2), wvb, aeb);

    gemm(wvb, wt_no[l], P(base + 8), anb, MNOD, DD, HD, 0);

    k_add_ln<<<dim3(BB), dim3(256), 0, stream>>>(anb, nodesA, nodesB, NN * DD);
    k_add_ln<<<dim3(BB), dim3(256), 0, stream>>>(aeb, edgesA, edgesB, NN * NN * DD);

    gemm(nodesB, wt_mn1[l], P(base + 26), anb, MNOD, DD, DD, 1);
    gemm(anb,    wt_mn2[l], P(base + 28), tnb, MNOD, DD, DD, 1);
    k_add_ln<<<dim3(BB), dim3(256), 0, stream>>>(tnb, nodesB, nodesA, NN * DD);

    gemm(edgesB, wt_me1[l], P(base + 22), aeb, MEDG, DD, DD, 1);
    gemm(aeb,    wt_me2[l], P(base + 24), teb, MEDG, DD, DD, 1);
    k_add_ln<<<dim3(BB), dim3(256), 0, stream>>>(teb, edgesB, edgesA, NN * NN * DD);
  }

  // ---- heads: g = nodesA viewed as [64][1024] ----
  float* outp = (float*)d_out;
  gemm(nodesA, wt_mm1, P(70), tnb,  BB, 512, 1024, 1);
  gemm(tnb,    wt_mm2, P(72), outp, BB, 256, 512,  1);
  gemm(nodesA, wt_mv1, P(74), tnb,  BB, 512, 1024, 1);
  gemm(tnb,    wt_mv2, P(76), outp + (size_t)BB * 256, BB, 256, 512, 1);
}
